// GAT_pyg_17119739641948
// MI455X (gfx1250) — compile-verified, matrix operand order checked by reading, not run on hardware
//
#include <hip/hip_runtime.h>
#include <hip/hip_bf16.h>

// ---------------------------------------------------------------------------
// Types for CDNA5 WMMA (wave32): v_wmma_f32_16x16x32_bf16
// ---------------------------------------------------------------------------
typedef __attribute__((__ext_vector_type__(16))) __bf16 bf16x16;
typedef __attribute__((__ext_vector_type__(8)))  float  f32x8;
typedef __attribute__((__ext_vector_type__(4)))  unsigned int u32x4;

union Frag16 {                // 16 bf16 = 8 VGPRs
    u32x4   q[2];
    bf16x16 v;
};

// float -> bf16, round-to-nearest-even
static __device__ __forceinline__ unsigned short f2bf(float f) {
    unsigned u = __float_as_uint(f);
    u += 0x7FFFu + ((u >> 16) & 1u);
    return (unsigned short)(u >> 16);
}

// ordered-uint encoding so unsigned atomicMax == float max
static __device__ __forceinline__ unsigned fkey(float f) {
    unsigned u = __float_as_uint(f);
    return (u & 0x80000000u) ? ~u : (u | 0x80000000u);
}
static __device__ __forceinline__ float funkey(unsigned k) {
    unsigned u = (k & 0x80000000u) ? (k & 0x7FFFFFFFu) : ~k;
    return __uint_as_float(u);
}

// ---------------------------------------------------------------------------
// bf16 WMMA GEMM:  Y[M,NOUT] (f32) = A[M,K] (bf16, row-major) @ Bt[NOUT,K]^T
// Bt is the transposed weight so both fragments load contiguously.
// 4 waves per block, each wave owns one 16-row tile, sweeps all NOUT columns.
// ---------------------------------------------------------------------------
#define GEMM_WAVES 4
__global__ void wmma_gemm_bf16(const unsigned short* __restrict__ A,
                               const unsigned short* __restrict__ Bt,
                               float* __restrict__ Y,
                               int M, int K, int NOUT)
{
    const int lane  = threadIdx.x & 31;
    const int wave  = threadIdx.x >> 5;
    const int tileM = blockIdx.x * GEMM_WAVES + wave;   // wave-uniform
    const int m0    = tileM * 16;
    if (m0 >= M) return;                                // whole-wave exit

    const int l15   = lane & 15;
    const int half  = lane >> 4;                        // 0 | 1
    const int kbase = half * 8;

    int rowA = m0 + l15;
    if (rowA >= M) rowA = M - 1;                        // clamp; stores guarded
    const unsigned short* aptr = A + (size_t)rowA * K + kbase;

    for (int c0 = 0; c0 < NOUT; c0 += 16) {
        const unsigned short* bptr = Bt + (size_t)(c0 + l15) * K + kbase;
        f32x8 acc = {};
        for (int k0 = 0; k0 < K; k0 += 32) {
            Frag16 fa, fb;
            fa.q[0] = *(const u32x4*)(aptr + k0);        // K = kbase .. +7
            fa.q[1] = *(const u32x4*)(aptr + k0 + 16);   // K = kbase+16 .. +23
            fb.q[0] = *(const u32x4*)(bptr + k0);
            fb.q[1] = *(const u32x4*)(bptr + k0 + 16);
            acc = __builtin_amdgcn_wmma_f32_16x16x32_bf16(
                      false, fa.v, false, fb.v, (short)0, acc, false, false);
        }
        const int colY = c0 + l15;
#pragma unroll
        for (int r = 0; r < 8; ++r) {
            const int row = m0 + r + half * 8;
            if (row < M) Y[(size_t)row * NOUT + colY] = acc[r];
        }
    }
}

// ---------------------------------------------------------------------------
// es[i] = h[i,:].a_s ; ed[i] = h[i,:].a_d   (one wave per node)
// ---------------------------------------------------------------------------
__global__ void node_scores(const float* __restrict__ h,
                            const float* __restrict__ as_, const float* __restrict__ ad_,
                            float* __restrict__ es, float* __restrict__ ed,
                            int N, int C)
{
    const int warp = (blockIdx.x * blockDim.x + threadIdx.x) >> 5;
    const int lane = threadIdx.x & 31;
    if (warp >= N) return;
    const float* row = h + (size_t)warp * C;
    float s = 0.f, d = 0.f;
    for (int c = lane; c < C; c += 32) {
        const float v = row[c];
        s += v * as_[c];
        d += v * ad_[c];
    }
#pragma unroll
    for (int off = 16; off; off >>= 1) {
        s += __shfl_xor(s, off);
        d += __shfl_xor(d, off);
    }
    if (lane == 0) { es[warp] = s; ed[warp] = d; }
}

__global__ void init_node(unsigned* __restrict__ maxbits, float* __restrict__ denom, int N)
{
    const int i = blockIdx.x * blockDim.x + threadIdx.x;
    if (i >= N) return;
    maxbits[i] = fkey(-__builtin_inff());
    denom[i]   = 0.f;
}

__global__ void init_agg(float* __restrict__ agg, const float* __restrict__ bias, int N, int C)
{
    const size_t i = (size_t)blockIdx.x * blockDim.x + threadIdx.x;
    if (i >= (size_t)N * C) return;
    agg[i] = bias[i % C];
}

// ---------------------------------------------------------------------------
// Edge passes (edge e in [0, E+N); e >= E is the self-loop of node e-E)
// ---------------------------------------------------------------------------
__global__ void edge_logit_max(const int* __restrict__ ei, int E, int N,
                               const float* __restrict__ es, const float* __restrict__ ed,
                               float* __restrict__ logits, unsigned* __restrict__ maxbits)
{
    const int e = blockIdx.x * blockDim.x + threadIdx.x;
    if (e >= E + N) return;
    const int s = (e < E) ? ei[e]     : e - E;
    const int d = (e < E) ? ei[E + e] : e - E;
    float l = es[s] + ed[d];
    l = (l > 0.f) ? l : 0.2f * l;                 // LeakyReLU(0.2)
    logits[e] = l;
    atomicMax(&maxbits[d], fkey(l));
}

__global__ void edge_exp(const int* __restrict__ ei, int E, int N,
                         const float* __restrict__ logits, const unsigned* __restrict__ maxbits,
                         float* __restrict__ p, float* __restrict__ denom)
{
    const int e = blockIdx.x * blockDim.x + threadIdx.x;
    if (e >= E + N) return;
    const int d = (e < E) ? ei[E + e] : e - E;
    const float v = __expf(logits[e] - funkey(maxbits[d]));
    p[e] = v;
    atomicAdd(&denom[d], v);
}

__global__ void edge_aggregate(const int* __restrict__ ei, int E, int N,
                               const float* __restrict__ h, const float* __restrict__ p,
                               const float* __restrict__ denom, float* __restrict__ agg, int C)
{
    const int NE = E + N;
    for (int e = blockIdx.x; e < NE; e += gridDim.x) {
        const int s = (e < E) ? ei[e]     : e - E;
        const int d = (e < E) ? ei[E + e] : e - E;
        const float alpha = p[e] / denom[d];
        const float* hs = h   + (size_t)s * C;
        float*       od = agg + (size_t)d * C;
        for (int c = threadIdx.x; c < C; c += blockDim.x)
            atomicAdd(&od[c], hs[c] * alpha);
    }
}

// ---------------------------------------------------------------------------
// Elementwise conversions
// ---------------------------------------------------------------------------
__global__ void to_bf16(const float* __restrict__ x, unsigned short* __restrict__ y, size_t n)
{
    const size_t i = (size_t)blockIdx.x * blockDim.x + threadIdx.x;
    if (i < n) y[i] = f2bf(x[i]);
}

__global__ void relu_to_bf16(const float* __restrict__ x, unsigned short* __restrict__ y, size_t n)
{
    const size_t i = (size_t)blockIdx.x * blockDim.x + threadIdx.x;
    if (i < n) { const float v = x[i]; y[i] = f2bf(v > 0.f ? v : 0.f); }
}

// W[k][n] (K x NOUT, row-major) -> Wt[n][k] bf16
__global__ void transpose_to_bf16(const float* __restrict__ W, unsigned short* __restrict__ Wt,
                                  int K, int NOUT)
{
    const int i = blockIdx.x * blockDim.x + threadIdx.x;
    if (i >= K * NOUT) return;
    const int k = i / NOUT, n = i % NOUT;
    Wt[(size_t)n * K + k] = f2bf(W[i]);
}

// ---------------------------------------------------------------------------
// Final link predictor: sigmoid( [h[m0], h[m1]] . Wl + bl ), one wave per pair
// ---------------------------------------------------------------------------
__global__ void link_predict(const float* __restrict__ h, const int* __restrict__ mask,
                             const float* __restrict__ Wl, const float* __restrict__ bl,
                             float* __restrict__ out, int P, int C)
{
    const int warp = (blockIdx.x * blockDim.x + threadIdx.x) >> 5;
    const int lane = threadIdx.x & 31;
    if (warp >= P) return;
    const int m0 = mask[warp * 2 + 0];
    const int m1 = mask[warp * 2 + 1];
    const float* r0 = h + (size_t)m0 * C;
    const float* r1 = h + (size_t)m1 * C;
    float acc = 0.f;
    for (int c = lane; c < C; c += 32)
        acc += r0[c] * Wl[c] + r1[c] * Wl[C + c];
#pragma unroll
    for (int off = 16; off; off >>= 1) acc += __shfl_xor(acc, off);
    if (lane == 0) out[warp] = 1.f / (1.f + expf(-(acc + bl[0])));
}

// ---------------------------------------------------------------------------
// Host orchestration
// ---------------------------------------------------------------------------
extern "C" void kernel_launch(void* const* d_in, const int* in_sizes, int n_in,
                              void* d_out, int out_size, void* d_ws, size_t ws_size,
                              hipStream_t stream)
{
    (void)n_in; (void)out_size; (void)ws_size;

    const float* features = (const float*)d_in[0];
    const int*   ei       = (const int*)  d_in[1];
    const int*   mask     = (const int*)  d_in[2];
    const float* W1   = (const float*)d_in[3];
    const float* a_s1 = (const float*)d_in[4];
    const float* a_d1 = (const float*)d_in[5];
    const float* b1   = (const float*)d_in[6];
    const float* W2   = (const float*)d_in[7];
    const float* a_s2 = (const float*)d_in[8];
    const float* a_d2 = (const float*)d_in[9];
    const float* b2   = (const float*)d_in[10];
    const float* Wl   = (const float*)d_in[11];
    const float* bl   = (const float*)d_in[12];

    const int F  = 128, Hh = 256;
    const int N  = in_sizes[0] / F;
    const int E  = in_sizes[1] / 2;
    const int P  = in_sizes[2] / 2;
    const int NE = E + N;

    // bump allocator over d_ws (~146 MB total)
    char* ws = (char*)d_ws;
    size_t off = 0;
    auto alloc = [&](size_t bytes) -> char* {
        char* ptr = ws + off;
        off += (bytes + 255) & ~(size_t)255;
        return ptr;
    };
    unsigned short* xbf  = (unsigned short*)alloc((size_t)N * F  * 2);  // bf16 features
    unsigned short* w1t  = (unsigned short*)alloc((size_t)Hh * F * 2);  // W1^T bf16
    unsigned short* w2t  = (unsigned short*)alloc((size_t)F * Hh * 2);  // W2^T bf16
    unsigned short* hrbf = (unsigned short*)alloc((size_t)N * Hh * 2);  // relu(h1) bf16
    float*    buf1    = (float*)   alloc((size_t)N * Hh * 4);           // h1, then h2
    float*    buf2    = (float*)   alloc((size_t)N * Hh * 4);           // agg1, then agg2
    float*    es      = (float*)   alloc((size_t)N * 4);
    float*    ed      = (float*)   alloc((size_t)N * 4);
    unsigned* maxbits = (unsigned*)alloc((size_t)N * 4);
    float*    denom   = (float*)   alloc((size_t)N * 4);
    float*    logits  = (float*)   alloc((size_t)NE * 4);
    float*    pbuf    = (float*)   alloc((size_t)NE * 4);

    const int tilesM = (N + 15) / 16;
    const int gemmGrid = (tilesM + GEMM_WAVES - 1) / GEMM_WAVES;
    const int TB = 256;
    auto cdiv = [](size_t a, size_t b) -> int { return (int)((a + b - 1) / b); };

    // --- conversions -------------------------------------------------------
    to_bf16<<<cdiv((size_t)N * F, TB), TB, 0, stream>>>(features, xbf, (size_t)N * F);
    transpose_to_bf16<<<cdiv((size_t)F * Hh, TB), TB, 0, stream>>>(W1, w1t, F, Hh);
    transpose_to_bf16<<<cdiv((size_t)Hh * F, TB), TB, 0, stream>>>(W2, w2t, Hh, F);

    // --- layer 1 -----------------------------------------------------------
    wmma_gemm_bf16<<<gemmGrid, 32 * GEMM_WAVES, 0, stream>>>(xbf, w1t, buf1, N, F, Hh);
    node_scores<<<cdiv((size_t)N * 32, TB), TB, 0, stream>>>(buf1, a_s1, a_d1, es, ed, N, Hh);
    init_node<<<cdiv(N, TB), TB, 0, stream>>>(maxbits, denom, N);
    init_agg<<<cdiv((size_t)N * Hh, TB), TB, 0, stream>>>(buf2, b1, N, Hh);
    edge_logit_max<<<cdiv(NE, TB), TB, 0, stream>>>(ei, E, N, es, ed, logits, maxbits);
    edge_exp<<<cdiv(NE, TB), TB, 0, stream>>>(ei, E, N, logits, maxbits, pbuf, denom);
    edge_aggregate<<<8192, Hh, 0, stream>>>(ei, E, N, buf1, pbuf, denom, buf2, Hh);
    relu_to_bf16<<<cdiv((size_t)N * Hh, TB), TB, 0, stream>>>(buf2, hrbf, (size_t)N * Hh);

    // --- layer 2 (h2 overwrites buf1; agg2 overwrites buf2) ---------------
    wmma_gemm_bf16<<<gemmGrid, 32 * GEMM_WAVES, 0, stream>>>(hrbf, w2t, buf1, N, Hh, F);
    node_scores<<<cdiv((size_t)N * 32, TB), TB, 0, stream>>>(buf1, a_s2, a_d2, es, ed, N, F);
    init_node<<<cdiv(N, TB), TB, 0, stream>>>(maxbits, denom, N);
    init_agg<<<cdiv((size_t)N * F, TB), TB, 0, stream>>>(buf2, b2, N, F);
    edge_logit_max<<<cdiv(NE, TB), TB, 0, stream>>>(ei, E, N, es, ed, logits, maxbits);
    edge_exp<<<cdiv(NE, TB), TB, 0, stream>>>(ei, E, N, logits, maxbits, pbuf, denom);
    edge_aggregate<<<8192, F, 0, stream>>>(ei, E, N, buf1, pbuf, denom, buf2, F);

    // --- link prediction ---------------------------------------------------
    link_predict<<<cdiv((size_t)P * 32, TB), TB, 0, stream>>>(buf2, mask, Wl, bl,
                                                              (float*)d_out, P, F);
}
